// ST_GCN_7421703487970
// MI455X (gfx1250) — compile-verified, matrix operand order checked by reading, not run on hardware
//
#include <hip/hip_runtime.h>
#include <hip/hip_bf16.h>

// ST-GCN for MI455X (gfx1250, wave32, WMMA f32_16x16x32_f16)
// B=32, T=64, N=512, H=64, C1=32, BT=2048

#define Bn 32
#define Tn 64
#define Nn 512
#define Hn 64
#define C1n 32
#define BTn 2048

typedef __attribute__((ext_vector_type(16))) _Float16 v16h;
typedef __attribute__((ext_vector_type(8)))  _Float16 v8h;
typedef __attribute__((ext_vector_type(8)))  float    v8f;

#define WMMA_F16(a, b, c) \
  __builtin_amdgcn_wmma_f32_16x16x32_f16(false, (a), false, (b), (short)0, (c), false, false)

// Load one 16x32 f16 fragment (A layout, or B^T with identical striping) from
// a row-major matrix. rowbase = matrix + row*ld (row = M for A, N for B^T).
// Per ISA: lane's 16 halfs = [k0 + half*8 .. +7] and [k0+16+half*8 .. +7].
__device__ __forceinline__ v16h load_frag_row(const _Float16* __restrict__ rowbase,
                                              int k0, int halfsel) {
  v8h lo = *(const v8h*)(rowbase + k0 + halfsel * 8);
  v8h hi = *(const v8h*)(rowbase + k0 + halfsel * 8 + 16);
  v16h r;
#pragma unroll
  for (int i = 0; i < 8; ++i) { r[i] = lo[i]; r[i + 8] = hi[i]; }
  return r;
}

// Same fragment gather but from LDS (base already includes halfsel*8 offset).
__device__ __forceinline__ v16h lds_frag(const _Float16* base) {
  v8h lo = *(const v8h*)(base);
  v8h hi = *(const v8h*)(base + 16);
  v16h r;
#pragma unroll
  for (int i = 0; i < 8; ++i) { r[i] = lo[i]; r[i + 8] = hi[i]; }
  return r;
}

// ---------------- conversion kernels ----------------
__global__ __launch_bounds__(256) void k_cvt_adj(const float* __restrict__ adj,
                                                 _Float16* __restrict__ adj_h) {
  int i = blockIdx.x * 256 + threadIdx.x;
  if (i < Nn * Nn) adj_h[i] = (_Float16)adj[i];
}

__global__ __launch_bounds__(256) void k_cvt_x(const float* __restrict__ x,
                                               _Float16* __restrict__ xh) {
  int i = blockIdx.x * 256 + threadIdx.x;
  if (i < BTn * Nn) xh[i] = (_Float16)x[i];   // natural (bt, v) layout = B^T for GEMM1
}

// tcn_w (N*H, H, 3) -> tw[n][k][ho][hi]  (B^T layout for the TCN GEMM)
__global__ __launch_bounds__(256) void k_cvt_tw(const float* __restrict__ tcn_w,
                                                _Float16* __restrict__ tw) {
  int idx = blockIdx.x * 256 + threadIdx.x;
  if (idx >= Nn * 3 * Hn * Hn) return;
  int hi = idx & 63;
  int ho = (idx >> 6) & 63;
  int nk = idx >> 12;
  int k = nk % 3;
  int n = nk / 3;
  tw[idx] = (_Float16)tcn_w[(size_t)(n * Hn + ho) * (Hn * 3) + hi * 3 + k];
}

// ---------------- GEMM1: Y[bt,u] = sum_v adj[u,v] * x[bt,v] ----------------
// 1024 waves: wave = ut(32) x g(32); fully unrolled, double-buffered K loop
// (full unroll -> SSA renaming kills all ping-pong register copies).
__global__ __launch_bounds__(256) void k_gemm1(const _Float16* __restrict__ adj_h,
                                               const _Float16* __restrict__ xh,
                                               float* __restrict__ y) {
  int lane = threadIdx.x & 31;
  int lanelo = lane & 15, halfsel = lane >> 4;
  int wave = blockIdx.x * 8 + (threadIdx.x >> 5);
  int ut = wave >> 5;
  int g = wave & 31;
  const _Float16* Arow = adj_h + (size_t)(ut * 16 + lanelo) * Nn;
  const _Float16* Brow = xh + (size_t)(g * 64 + lanelo) * Nn;
  v8f zero = {};
  v8f acc[4];
#pragma unroll
  for (int j = 0; j < 4; ++j) acc[j] = zero;

  v16h a = load_frag_row(Arow, 0, halfsel);
  v16h b[4];
#pragma unroll
  for (int j = 0; j < 4; ++j) b[j] = load_frag_row(Brow + (size_t)j * 16 * Nn, 0, halfsel);

#pragma unroll
  for (int s = 0; s < 16; ++s) {
    v16h an;
    v16h bn[4];
    if (s < 15) {
      int k0 = (s + 1) * 32;
      an = load_frag_row(Arow, k0, halfsel);
#pragma unroll
      for (int j = 0; j < 4; ++j) bn[j] = load_frag_row(Brow + (size_t)j * 16 * Nn, k0, halfsel);
    }
#pragma unroll
    for (int j = 0; j < 4; ++j) acc[j] = WMMA_F16(a, b[j], acc[j]);
    if (s < 15) {
      a = an;
#pragma unroll
      for (int j = 0; j < 4; ++j) b[j] = bn[j];
    }
  }
#pragma unroll
  for (int j = 0; j < 4; ++j) {
    int bt = g * 64 + j * 16 + lanelo;
#pragma unroll
    for (int r = 0; r < 8; ++r) {
      int u = ut * 16 + r + halfsel * 8;
      y[(size_t)bt * Nn + u] = acc[j][r];
    }
  }
}

// ---------------- fused h1 + G: G^T[bt][h*512+v] ----------------
__global__ __launch_bounds__(256) void k_h1g(const float* __restrict__ y,
                                             const float* __restrict__ w1,
                                             const float* __restrict__ b1,
                                             const float* __restrict__ w2,
                                             _Float16* __restrict__ ght) {
  __shared__ float w2s[C1n * Hn];
  __shared__ float w1s[C1n];
  __shared__ float b1s[C1n];
  for (int i = threadIdx.x; i < C1n * Hn; i += 256) w2s[i] = w2[i];
  if (threadIdx.x < C1n) { w1s[threadIdx.x] = w1[threadIdx.x]; b1s[threadIdx.x] = b1[threadIdx.x]; }
  __syncthreads();

  int tid = blockIdx.x * 256 + threadIdx.x;   // BTn * Nn threads
  int bt = tid >> 9;
  int v = tid & 511;
  float yv = y[(size_t)bt * Nn + v];
  float g[Hn];
#pragma unroll
  for (int h = 0; h < Hn; ++h) g[h] = 0.f;
  for (int c = 0; c < C1n; ++c) {
    float h1 = fmaxf(yv * w1s[c] + b1s[c], 0.f);
#pragma unroll
    for (int h = 0; h < Hn; ++h) g[h] += h1 * w2s[c * Hn + h];
  }
  _Float16* dst = ght + (size_t)bt * (Hn * Nn) + v;
#pragma unroll
  for (int h = 0; h < Hn; ++h) dst[(size_t)h * Nn] = (_Float16)g[h];
}

// ---------------- GEMM2 (dominant, 68.7 GFLOP) ----------------
// h2[bt,u,h] = relu(sum_v adj[u,v] G[bt,v,h] + b2[h]); output (n=u,b,t,h) f16.
// Block (8 waves) = one bt x 256-row u panel. B panel (64h x 32k) staged in LDS,
// double buffered; A (adj rows) register double-buffered; each wave: 32x64 tile.
// Fully unrolled K loop so buffer parity and ping-pong copies fold away.
#define PADK 40   // 32 + 8 halfs pad -> 80B row stride, conflict-free ds_load_b128
__global__ __launch_bounds__(256) void k_gemm2(const _Float16* __restrict__ adj_h,
                                               const _Float16* __restrict__ ght,
                                               const float* __restrict__ b2,
                                               _Float16* __restrict__ h2s) {
  __shared__ alignas(16) _Float16 Bs[2][64 * PADK];
  int tid = threadIdx.x;
  int lane = tid & 31, lanelo = lane & 15, halfsel = lane >> 4;
  int wv = tid >> 5;                   // 0..7
  int bt = blockIdx.x >> 1;
  int ubase = (blockIdx.x & 1) * 256 + wv * 32;   // wave: u rows [ubase, ubase+32)

  const _Float16* Bg = ght + (size_t)bt * (Hn * Nn);   // [h][v]
  int srow = tid >> 2;                 // staging: 4 threads per h-row
  int skoff = (tid & 3) * 8;           // 8 halfs (16B) each

  const _Float16* Arow0 = adj_h + (size_t)(ubase + lanelo) * Nn;
  const _Float16* Arow1 = Arow0 + (size_t)16 * Nn;

  v8f zero = {};
  v8f acc[8];
#pragma unroll
  for (int j = 0; j < 8; ++j) acc[j] = zero;

  // stage K-step 0
  *(v8h*)&Bs[0][srow * PADK + skoff] = *(const v8h*)(Bg + (size_t)srow * Nn + skoff);
  __syncthreads();
  v16h a0 = load_frag_row(Arow0, 0, halfsel);
  v16h a1 = load_frag_row(Arow1, 0, halfsel);

#pragma unroll
  for (int s = 0; s < 16; ++s) {
    int cur = s & 1;
    v8h stn;
    v16h a0n, a1n;
    if (s < 15) {                      // issue next-step global loads first
      int k0n = (s + 1) * 32;
      stn = *(const v8h*)(Bg + (size_t)srow * Nn + k0n + skoff);
      a0n = load_frag_row(Arow0, k0n, halfsel);
      a1n = load_frag_row(Arow1, k0n, halfsel);
    }
    v16h b[4];
#pragma unroll
    for (int j = 0; j < 4; ++j)
      b[j] = lds_frag(&Bs[cur][(j * 16 + lanelo) * PADK + halfsel * 8]);
#pragma unroll
    for (int j = 0; j < 4; ++j) {
      acc[j]     = WMMA_F16(a0, b[j], acc[j]);
      acc[j + 4] = WMMA_F16(a1, b[j], acc[j + 4]);
    }
    if (s < 15) {
      *(v8h*)&Bs[1 - cur][srow * PADK + skoff] = stn;
      a0 = a0n;
      a1 = a1n;
    }
    __syncthreads();
  }

  int bb = bt >> 6, t = bt & 63;
#pragma unroll
  for (int j = 0; j < 4; ++j) {
    int h = j * 16 + lanelo;
    float bias = b2[h];
#pragma unroll
    for (int p = 0; p < 2; ++p) {
#pragma unroll
      for (int r = 0; r < 8; ++r) {
        int u = ubase + p * 16 + r + halfsel * 8;
        float z = acc[j + p * 4][r] + bias;
        z = z > 0.f ? z : 0.f;
        h2s[(((size_t)u * Bn + bb) * Tn + t) * Hn + h] = (_Float16)z;
      }
    }
  }
}

// ---------------- out init ----------------
__global__ __launch_bounds__(256) void k_init_out(float* __restrict__ out,
                                                  const float* __restrict__ fc_b) {
  int i = blockIdx.x * 256 + threadIdx.x;
  if (i < Bn * Nn) out[i] = fc_b[0];
}

// ---------------- TCN (3 shifted GEMMs) fused with FC reduction ----------------
// 65536 waves: wave = n(512) x mt(128). All 8 waves of a block share node n:
// per-node weights (3x64x64) + fc_w + bias staged in LDS once per block.
#define PADW 72   // 64 + 8 halfs pad -> 144B row stride, conflict-free ds_load_b128
__global__ __launch_bounds__(256) void k_tcn_fc(const _Float16* __restrict__ h2s,
                                                const _Float16* __restrict__ tw,
                                                const float* __restrict__ tcn_b,
                                                const float* __restrict__ fc_w,
                                                float* __restrict__ out) {
  __shared__ alignas(16) _Float16 w_s[3 * 64 * PADW];
  __shared__ float fcw_s[Hn * Tn];
  __shared__ float bias_s[Hn];

  int lane = threadIdx.x & 31;
  int lanelo = lane & 15, halfsel = lane >> 4;
  int wave = blockIdx.x * 8 + (threadIdx.x >> 5);
  int n = wave >> 7;        // uniform within block (8 | 128)
  int mt = wave & 127;

  const _Float16* twn = tw + (size_t)n * (3 * Hn * Hn);
  for (int c = threadIdx.x; c < 192 * 8; c += 256) {   // 192 rows x 64 halfs
    int row = c >> 3, off = (c & 7) * 8;
    *(v8h*)&w_s[row * PADW + off] = *(const v8h*)(twn + (size_t)row * Hn + off);
  }
  for (int i = threadIdx.x; i < Hn * Tn; i += 256) fcw_s[i] = fc_w[i];
  if (threadIdx.x < Hn) bias_s[threadIdx.x] = tcn_b[n * Hn + threadIdx.x];
  __syncthreads();

  int row = mt * 16 + lanelo;   // bt' = b*T + t in this node's (2048 x 64) matrix
  int t = row & 63;
  int bbase = row & ~63;
  const _Float16* in = h2s + (size_t)n * (BTn * Hn);

  bool valid[3];
  const _Float16* Ar[3];
#pragma unroll
  for (int k = 0; k < 3; ++k) {
    int ts = t + k - 1;
    valid[k] = ((unsigned)ts) < (unsigned)Tn;
    Ar[k] = in + (size_t)(bbase + (valid[k] ? ts : 0)) * Hn;
  }

  v8f zero = {};
  v8f acc[4];
#pragma unroll
  for (int j = 0; j < 4; ++j) acc[j] = zero;

  // flat steps: step = k*2 + kk, k0 = kk*32; A register double-buffered, B from LDS
  v16h a = load_frag_row(Ar[0], 0, halfsel);
  if (!valid[0]) { v16h z = {}; a = z; }
#pragma unroll
  for (int step = 0; step < 6; ++step) {
    v16h an;
    if (step < 5) {
      int kn = (step + 1) >> 1, k0n = ((step + 1) & 1) * 32;
      an = load_frag_row(Ar[kn], k0n, halfsel);
      if (!valid[kn]) { v16h z = {}; an = z; }
    }
    int k = step >> 1, k0 = (step & 1) * 32;
    v16h b[4];
#pragma unroll
    for (int j = 0; j < 4; ++j)
      b[j] = lds_frag(&w_s[(k * 64 + j * 16 + lanelo) * PADW + k0 + halfsel * 8]);
#pragma unroll
    for (int j = 0; j < 4; ++j) acc[j] = WMMA_F16(a, b[j], acc[j]);
    if (step < 5) a = an;
  }

  float partial = 0.f;
#pragma unroll
  for (int j = 0; j < 4; ++j) {
    int ho = j * 16 + lanelo;
    float bias = bias_s[ho];
#pragma unroll
    for (int r = 0; r < 8; ++r) {
      int rr = mt * 16 + r + halfsel * 8;
      int tt = rr & 63;
      float v = acc[j][r] + bias;
      v = v > 0.f ? v : 0.f;
      partial += v * fcw_s[ho * Tn + tt];
    }
  }
#pragma unroll
  for (int off = 16; off > 0; off >>= 1)
    partial += __shfl_xor(partial, off, 32);
  if (lane == 0) {
    int bb = mt >> 2;   // 16-aligned row tiles never cross a b boundary (16 | 64)
    atomicAdd(&out[bb * Nn + n], partial);
  }
}

// ---------------- host launcher ----------------
extern "C" void kernel_launch(void* const* d_in, const int* in_sizes, int n_in,
                              void* d_out, int out_size, void* d_ws, size_t ws_size,
                              hipStream_t stream) {
  const float* x     = (const float*)d_in[0];
  const float* adj   = (const float*)d_in[1];
  const float* w1    = (const float*)d_in[2];
  const float* b1    = (const float*)d_in[3];
  const float* w2    = (const float*)d_in[4];
  const float* b2    = (const float*)d_in[5];
  const float* tcn_w = (const float*)d_in[6];
  const float* tcn_b = (const float*)d_in[7];
  const float* fc_w  = (const float*)d_in[8];
  const float* fc_b  = (const float*)d_in[9];
  float* out = (float*)d_out;

  const size_t off_adj = 0;                              // 512*512 f16      = 0.5 MiB
  const size_t off_xh  = off_adj + (size_t)Nn * Nn * 2;  // 2048*512 f16     = 2 MiB
  const size_t off_y   = off_xh + (size_t)BTn * Nn * 2;  // 2048*512 f32     = 4 MiB
  const size_t off_g   = off_y + (size_t)BTn * Nn * 4;   // 2048*64*512 f16  = 128 MiB
  const size_t off_h2  = off_g + (size_t)BTn * Hn * Nn * 2;  // same size    = 128 MiB
  const size_t off_tw  = off_h2 + (size_t)BTn * Hn * Nn * 2; // 512*3*64*64  = 12 MiB
  const size_t need    = off_tw + (size_t)Nn * 3 * Hn * Hn * 2;
  if (ws_size < need) return;

  char* ws = (char*)d_ws;
  _Float16* adj_h = (_Float16*)(ws + off_adj);
  _Float16* xh    = (_Float16*)(ws + off_xh);
  float*    y     = (float*)(ws + off_y);
  _Float16* ght   = (_Float16*)(ws + off_g);
  _Float16* h2s   = (_Float16*)(ws + off_h2);
  _Float16* tw    = (_Float16*)(ws + off_tw);

  k_cvt_adj<<<(Nn * Nn + 255) / 256, 256, 0, stream>>>(adj, adj_h);
  k_cvt_x<<<(BTn * Nn + 255) / 256, 256, 0, stream>>>(x, xh);
  k_cvt_tw<<<(Nn * 3 * Hn * Hn + 255) / 256, 256, 0, stream>>>(tcn_w, tw);

  k_gemm1<<<128, 256, 0, stream>>>(adj_h, xh, y);            // 1024 waves
  k_h1g<<<(BTn * Nn) / 256, 256, 0, stream>>>(y, w1, b1, w2, ght);
  k_gemm2<<<4096, 256, 0, stream>>>(adj_h, ght, b2, h2s);    // bt(2048) x u-half(2)
  k_init_out<<<(Bn * Nn + 255) / 256, 256, 0, stream>>>(out, fc_b);
  k_tcn_fc<<<8192, 256, 0, stream>>>(h2s, tw, tcn_b, fc_w, out); // 65536 waves
}
